// GRUClassifier_74543452389769
// MI455X (gfx1250) — compile-verified, matrix operand order checked by reading, not run on hardware
//
#include <hip/hip_runtime.h>

typedef _Float16 f16;
typedef __attribute__((ext_vector_type(8)))  _Float16 v8h;
typedef __attribute__((ext_vector_type(16))) _Float16 v16h;
typedef __attribute__((ext_vector_type(8)))  float    v8f;

#define B_   2048
#define T_   2048
#define I_   6
#define H_   64
#define G_   192      // 3*H
#define KAUG 96       // augmented K: 64 h + 6 x + 1 bias + 25 pad
#define STR  104      // LDS row stride (halves); 208 B, 16B-aligned
#define STRX 40       // LDS row stride for xn-B (halves); 80 B

__device__ __forceinline__ v16h join16(v8h lo, v8h hi) {
    return __builtin_shufflevector(lo, hi, 0,1,2,3,4,5,6,7,8,9,10,11,12,13,14,15);
}

// CDNA5 has a v_tanh_f32 transcendental; use it when the builtin exists.
#if __has_builtin(__builtin_amdgcn_tanhf)
__device__ __forceinline__ float tanh_fast(float x) { return __builtin_amdgcn_tanhf(x); }
#else
__device__ __forceinline__ float tanh_fast(float x) {
    // tanh(x) = 1 - 2/(exp(2x)+1), with raw v_rcp_f32 (no IEEE-div fixup chain)
    float e = __expf(2.0f * x);
    return 1.0f - 2.0f * __builtin_amdgcn_rcpf(e + 1.0f);
}
#endif
__device__ __forceinline__ float sig_fast(float x) {
    // sigmoid(x) = 0.5*tanh(x/2) + 0.5  -> one trans op + one fma
    return __builtin_fmaf(0.5f, tanh_fast(0.5f * x), 0.5f);
}

__global__ __launch_bounds__(128)
void gru_scan_kernel(const float* __restrict__ x,
                     const float* __restrict__ W_ih,
                     const float* __restrict__ W_hh,
                     const float* __restrict__ b_ih,
                     const float* __restrict__ b_hh,
                     const float* __restrict__ fc_w,
                     const float* __restrict__ fc_b,
                     float* __restrict__ out)
{
    // B-matrix staging: Bt[n][k], n = gate column (0..191), k = augmented K (0..95)
    __shared__ __attribute__((aligned(16))) f16  Bt [G_][STR];
    // xn-gate B (covers augmented K 64..95 only): BtX[j][k'], j = hidden idx
    __shared__ __attribute__((aligned(16))) f16  BtX[H_][STRX];
    // augmented A staging, double-buffered: hA[p][m][k] = [h | x_t | 1 | 0]
    __shared__ __attribute__((aligned(16))) f16  hA [2][16][STR];
    // final h for the FC head
    __shared__ __attribute__((aligned(16))) float hF[16][H_];

    const int tid  = threadIdx.x;
    const int lane = tid & 31;
    const int wv   = tid >> 5;           // wave 0..3 -> hidden cols 16w..16w+15
    const int col  = lane & 15;          // N within tile
    const int kh   = (lane >> 4) * 8;    // half-wave K offset (A/B frag layout)
    const int rb0  = blockIdx.x * 16;    // batch tile base

    // ---------------- one-time setup: build f16 B matrices in LDS ----------------
    for (int idx = tid; idx < G_ * KAUG; idx += 128) {
        int n = idx / KAUG, k = idx % KAUG;
        float v = 0.0f;
        if (k < 64)                      v = W_hh[n * H_ + k];
        else if (k < 70) { if (n < 128)  v = W_ih[n * I_ + (k - 64)]; }   // x part: r,z only
        else if (k == 70)                v = (n < 128) ? (b_ih[n] + b_hh[n]) : b_hh[n];
        Bt[n][k] = (f16)v;
    }
    for (int idx = tid; idx < H_ * 32; idx += 128) {
        int j = idx / 32, k = idx % 32;
        float v = 0.0f;
        if (k < 6)       v = W_ih[(128 + j) * I_ + k];
        else if (k == 6) v = b_ih[128 + j];
        BtX[j][k] = (f16)v;
    }
    for (int idx = tid; idx < 2 * 16 * KAUG; idx += 128) {
        int p = idx / (16 * KAUG);
        int r = idx - p * 16 * KAUG;
        int m = r / KAUG, k = r % KAUG;
        hA[p][m][k] = (k == 70) ? (f16)1.0f : (f16)0.0f;  // h0 = 0, bias-one, pad
    }
    __syncthreads();

    // ---------------- persistent B fragments in VGPRs ----------------
    v16h Br[3], Bz[3], Bn[3], Bx;
    {
        const int nr = 0 * H_ + 16 * wv + col;
        const int nz = 1 * H_ + 16 * wv + col;
        const int nn = 2 * H_ + 16 * wv + col;
        #pragma unroll
        for (int c = 0; c < 3; ++c) {
            const int kb = 32 * c + kh;
            Br[c] = join16(*(const v8h*)&Bt[nr][kb], *(const v8h*)&Bt[nr][kb + 16]);
            Bz[c] = join16(*(const v8h*)&Bt[nz][kb], *(const v8h*)&Bt[nz][kb + 16]);
            Bn[c] = join16(*(const v8h*)&Bt[nn][kb], *(const v8h*)&Bt[nn][kb + 16]);
        }
        const int j = 16 * wv + col;
        Bx = join16(*(const v8h*)&BtX[j][kh], *(const v8h*)&BtX[j][kh + 16]);
    }

    float h[8];
    #pragma unroll
    for (int v = 0; v < 8; ++v) h[v] = 0.0f;

    const int jcol = 16 * wv + col;       // hidden index this lane owns
    const int mrow = (lane >> 4) * 8;     // C/D row base
    const int arow = col;                 // A-fragment row = lane%16

    // x fetch assignment: threads 0..95 cover 16 rows x 6 inputs
    const int    xm    = tid / I_;
    const int    xi    = tid - xm * I_;
    const bool   isx   = (tid < 16 * I_);
    const size_t xrow  = (size_t)(rb0 + ((xm < 16) ? xm : 0)) * T_ * I_;

    // software pipeline: xv holds x[., t, .] when entering iteration t
    float xv = isx ? x[xrow + xi] : 0.0f;

    // ---------------- recurrent scan (one barrier per step) ----------------
    for (int t = 0; t < T_; ++t) {
        const int p = t & 1;
        // stage h (f16) into augmented A, ping-pong buffer
        #pragma unroll
        for (int v = 0; v < 8; ++v)
            hA[p][mrow + v][jcol] = (f16)h[v];
        // stage x_t (prefetched last iteration)
        if (isx) hA[p][xm][64 + xi] = (f16)xv;
        __syncthreads();

        // kick off next step's x load; latency overlaps WMMA + update below
        if (isx && (t + 1) < T_) {
            xv = x[xrow + (size_t)(t + 1) * I_ + xi];
            if (tid < 16 && ((t & 7) == 0) && (t + 40) < T_)
                __builtin_prefetch(&x[xrow + (size_t)(t + 40) * I_], 0, 1);
        }

        // A fragments (two ds_load_b128 each)
        v16h a0 = join16(*(const v8h*)&hA[p][arow][ 0 + kh], *(const v8h*)&hA[p][arow][ 0 + kh + 16]);
        v16h a1 = join16(*(const v8h*)&hA[p][arow][32 + kh], *(const v8h*)&hA[p][arow][32 + kh + 16]);
        v16h a2 = join16(*(const v8h*)&hA[p][arow][64 + kh], *(const v8h*)&hA[p][arow][64 + kh + 16]);

        v8f accR = {}, accZ = {}, accN = {}, accX = {};
        accR = __builtin_amdgcn_wmma_f32_16x16x32_f16(false, a0, false, Br[0], (short)0, accR, false, false);
        accZ = __builtin_amdgcn_wmma_f32_16x16x32_f16(false, a0, false, Bz[0], (short)0, accZ, false, false);
        accN = __builtin_amdgcn_wmma_f32_16x16x32_f16(false, a0, false, Bn[0], (short)0, accN, false, false);
        accR = __builtin_amdgcn_wmma_f32_16x16x32_f16(false, a1, false, Br[1], (short)0, accR, false, false);
        accZ = __builtin_amdgcn_wmma_f32_16x16x32_f16(false, a1, false, Bz[1], (short)0, accZ, false, false);
        accN = __builtin_amdgcn_wmma_f32_16x16x32_f16(false, a1, false, Bn[1], (short)0, accN, false, false);
        accR = __builtin_amdgcn_wmma_f32_16x16x32_f16(false, a2, false, Br[2], (short)0, accR, false, false);
        accZ = __builtin_amdgcn_wmma_f32_16x16x32_f16(false, a2, false, Bz[2], (short)0, accZ, false, false);
        accN = __builtin_amdgcn_wmma_f32_16x16x32_f16(false, a2, false, Bn[2], (short)0, accN, false, false);
        accX = __builtin_amdgcn_wmma_f32_16x16x32_f16(false, a2, false, Bx,    (short)0, accX, false, false);

        // gate nonlinearities + state update (per-lane; v_tanh_f32 fast path)
        #pragma unroll
        for (int v = 0; v < 8; ++v) {
            float r = sig_fast(accR[v]);
            float z = sig_fast(accZ[v]);
            float n = tanh_fast(__builtin_fmaf(r, accN[v], accX[v]));
            h[v] = __builtin_fmaf(z, h[v] - n, n);   // (1-z)*n + z*h
        }
        // no second barrier: ping-pong buffer + barrier at top of next iter
    }

    // ---------------- FC head: sigmoid(h @ fc_w^T + fc_b) ----------------
    __syncthreads();   // make sure last iteration's LDS reads are done before reuse
    #pragma unroll
    for (int v = 0; v < 8; ++v)
        hF[mrow + v][jcol] = h[v];
    __syncthreads();
    if (tid < 16) {
        float s = fc_b[0];
        #pragma unroll 8
        for (int j = 0; j < H_; ++j) s += hF[tid][j] * fc_w[j];
        out[rb0 + tid] = sig_fast(s);
    }
}

extern "C" void kernel_launch(void* const* d_in, const int* in_sizes, int n_in,
                              void* d_out, int out_size, void* d_ws, size_t ws_size,
                              hipStream_t stream) {
    (void)in_sizes; (void)n_in; (void)out_size; (void)d_ws; (void)ws_size;
    const float* x    = (const float*)d_in[0];
    const float* W_ih = (const float*)d_in[1];
    const float* W_hh = (const float*)d_in[2];
    const float* b_ih = (const float*)d_in[3];
    const float* b_hh = (const float*)d_in[4];
    const float* fc_w = (const float*)d_in[5];
    const float* fc_b = (const float*)d_in[6];
    float* out = (float*)d_out;

    dim3 grid(B_ / 16);   // 128 workgroups of 16 sequences
    dim3 block(128);      // 4 waves; wave w owns hidden cols 16w..16w+15
    gru_scan_kernel<<<grid, block, 0, stream>>>(x, W_ih, W_hh, b_ih, b_hh, fc_w, fc_b, out);
}
